// MM_model_73469710565562
// MI455X (gfx1250) — matrix operand order repaired and checked by reading, not compiled, not run on hardware
//
#include <hip/hip_runtime.h>
#include <hip/hip_bf16.h>

// ---------------------------------------------------------------------------
// CDNA5 (gfx1250, wave32) implementation of the multimodal GAT reference.
// GEMMs: V_WMMA_F32_16X16X32_BF16 (bf16x3 split precision) with
// Tensor-Data-Mover (TENSOR_LOAD_TO_LDS) staging of f32 tiles, pipelined
// one tile ahead of the WMMA compute.
// ---------------------------------------------------------------------------

typedef __attribute__((ext_vector_type(16))) __bf16    v16bf;
typedef __attribute__((ext_vector_type(8)))  float     v8f;
typedef __attribute__((ext_vector_type(4)))  unsigned  v4u;
typedef __attribute__((ext_vector_type(8)))  int       v8i;
typedef __attribute__((ext_vector_type(4)))  int       v4i;

#define DIMD 128

// ---- float <-> monotonic uint key (for exact atomic float max) ------------
__device__ __forceinline__ unsigned f2key(float f) {
    unsigned i = __float_as_uint(f);
    return (i & 0x80000000u) ? ~i : (i | 0x80000000u);
}
__device__ __forceinline__ float key2f(unsigned k) {
    unsigned i = (k & 0x80000000u) ? (k & 0x7fffffffu) : ~k;
    return __uint_as_float(i);
}
#define KEY_NEG_INF 0x007FFFFFu  // f2key(-inf)

// ---------------------------------------------------------------------------
// TDM: issue a 2D tile load (f32) global -> LDS.
//   tensor dims clamp/zero-pad the tail (OOB reads return 0).
//   row_stride in elements. tile is tile_d1 rows x tile_d0 elems, row-major
//   packed into LDS at lds_off.
// ---------------------------------------------------------------------------
__device__ __forceinline__ void tdm_load_2d_f32(const float* gptr, unsigned lds_off,
                                                int tensor_d0, int tensor_d1,
                                                long long row_stride,
                                                int tile_d0, int tile_d1)
{
    unsigned long long ga = (unsigned long long)gptr;
    unsigned long long s0 = (unsigned long long)row_stride;
    v4u g0;
    g0[0] = 1u;                                   // count=1, user descriptor
    g0[1] = lds_off;                              // lds_addr (bytes)
    g0[2] = (unsigned)ga;                         // global_addr[31:0]
    g0[3] = ((unsigned)(ga >> 32) & 0x01FFFFFFu)  // global_addr[56:32]
            | 0x80000000u;                        // type = 2 ("image")
    v8i g1;
    g1[0] = (int)(2u << 16);                      // data_size = 4 bytes
    g1[1] = (int)(((unsigned)tensor_d0 & 0xFFFFu) << 16);
    g1[2] = (int)((((unsigned)tensor_d0 >> 16) & 0xFFFFu)
                  | (((unsigned)tensor_d1 & 0xFFFFu) << 16));
    g1[3] = (int)((((unsigned)tensor_d1 >> 16) & 0xFFFFu)
                  | (((unsigned)tile_d0 & 0xFFFFu) << 16));
    g1[4] = (int)((unsigned)tile_d1 & 0xFFFFu);   // tile_dim1 | tile_dim2=0
    g1[5] = (int)(unsigned)(s0 & 0xFFFFFFFFull);  // tensor_dim0_stride[31:0]
    g1[6] = (int)(unsigned)((s0 >> 32) & 0xFFFFull); // stride[47:32] | d1s[15:0]=0
    g1[7] = 0;
    v4i z4 = {};
#if __clang_major__ >= 23
    v8i z8 = {};
    __builtin_amdgcn_tensor_load_to_lds(g0, g1, z4, z4, z8, 0);
#else
    __builtin_amdgcn_tensor_load_to_lds(g0, g1, z4, z4, 0);
#endif
}

// ---------------------------------------------------------------------------
// Tiled GEMM: C[N,128] (+)= A[N,K] * B[128,K]^T (+ bias), bf16x3 via WMMA.
// One workgroup = 128 rows x 128 cols; 8 waves, wave w owns M-tile w.
// TDM stages f32 tiles into LDS; waves convert to bf16 hi/lo; WMMA of tile k
// overlaps the TDM fetch of tile k+1.
// ---------------------------------------------------------------------------
__global__ __launch_bounds__(256)
void gemm_bf16x3_kernel(const float* __restrict__ A, int ldA,
                        const float* __restrict__ B, int ldB,
                        const float* __restrict__ bias,
                        float* __restrict__ C,
                        int N, int K, int accum)
{
    __shared__ float  sAf32[128 * 32];      // TDM staging, A tile (16KB)
    __shared__ float  sBf32[128 * 32];      // TDM staging, B tile (16KB)
    __shared__ __bf16 sAhi[128 * 32];
    __shared__ __bf16 sAlo[128 * 32];
    __shared__ __bf16 sBhi[128 * 32];
    __shared__ __bf16 sBlo[128 * 32];

    const int tid     = threadIdx.x;
    const int wave    = tid >> 5;
    const int lane    = tid & 31;
    const int lane15  = lane & 15;
    const int laneHi  = lane >> 4;          // 0 or 1
    const int rowBase = blockIdx.x * 128;

    v8f acc[8] = {};
    if (accum) {
        #pragma unroll
        for (int j = 0; j < 8; ++j) {
            #pragma unroll
            for (int r = 0; r < 8; ++r) {
                int row = rowBase + wave * 16 + r + laneHi * 8;
                int col = j * 16 + lane15;
                if (row < N) acc[j][r] = C[(size_t)row * DIMD + col];
            }
        }
    }

    const unsigned ldsA = (unsigned)(size_t)&sAf32[0];
    const unsigned ldsB = (unsigned)(size_t)&sBf32[0];

    // prologue: TDM fetch of tile 0 (wave 0 issues; EXEC ignored by TDM)
    if (tid < 32) {
        tdm_load_2d_f32(A + (size_t)rowBase * ldA, ldsA,
                        K, N - rowBase, ldA, 32, 128);
        tdm_load_2d_f32(B, ldsB, K, 128, ldB, 32, 128);
    }

    for (int k0 = 0; k0 < K; k0 += 32) {
        __builtin_amdgcn_s_wait_tensorcnt(0);   // f32 tile k0 landed in LDS
        __syncthreads();                         // ..and prior WMMA readers done

        // ---- convert staged f32 tile -> bf16 hi/lo tiles (LDS->LDS) ----
        #pragma unroll
        for (int i = 0; i < 16; ++i) {
            int idx = tid + i * 256;             // 0..4095
            float av = sAf32[idx];
            __bf16 ah = (__bf16)av;
            sAhi[idx] = ah;
            sAlo[idx] = (__bf16)(av - (float)ah);
            float bv = sBf32[idx];
            __bf16 bh = (__bf16)bv;
            sBhi[idx] = bh;
            sBlo[idx] = (__bf16)(bv - (float)bh);
        }
        __syncthreads();                         // bf16 ready; f32 staging free

        // ---- overlap: TDM fetch of tile k0+32 while we run WMMAs ----
        if (k0 + 32 < K && tid < 32) {
            int kn = k0 + 32;
            tdm_load_2d_f32(A + (size_t)rowBase * ldA + kn, ldsA,
                            K - kn, N - rowBase, ldA, 32, 128);
            tdm_load_2d_f32(B + kn, ldsB, K - kn, 128, ldB, 32, 128);
        }

        // ---- gather A fragment (16x32 bf16, CDNA5 wave32 layout) ----
        v16bf ahi, alo;
        {
            int mrow = wave * 16 + lane15;
            #pragma unroll
            for (int i = 0; i < 8; ++i) {
                int kk = ((i & 4) ? 16 : 0) + (laneHi ? 8 : 0) + 2 * (i & 3);
                ahi[2 * i]     = sAhi[mrow * 32 + kk];
                ahi[2 * i + 1] = sAhi[mrow * 32 + kk + 1];
                alo[2 * i]     = sAlo[mrow * 32 + kk];
                alo[2 * i + 1] = sAlo[mrow * 32 + kk + 1];
            }
        }

        #pragma unroll
        for (int j = 0; j < 8; ++j) {
            // ---- gather B fragment (32x16 bf16) ----
            v16bf bhi, blo;
            int ncol = j * 16 + lane15;
            #pragma unroll
            for (int i = 0; i < 8; ++i) {
                int kk = ((i & 4) ? 16 : 0) + (laneHi ? 8 : 0) + 2 * (i & 3);
                bhi[2 * i]     = sBhi[ncol * 32 + kk];
                bhi[2 * i + 1] = sBhi[ncol * 32 + kk + 1];
                blo[2 * i]     = sBlo[ncol * 32 + kk];
                blo[2 * i + 1] = sBlo[ncol * 32 + kk + 1];
            }
            // bf16x3: hi*hi + lo*hi + hi*lo, accumulated in f32
            acc[j] = __builtin_amdgcn_wmma_f32_16x16x32_bf16(
                         false, ahi, false, bhi, (short)0, acc[j], false, false);
            acc[j] = __builtin_amdgcn_wmma_f32_16x16x32_bf16(
                         false, alo, false, bhi, (short)0, acc[j], false, false);
            acc[j] = __builtin_amdgcn_wmma_f32_16x16x32_bf16(
                         false, ahi, false, blo, (short)0, acc[j], false, false);
        }
    }

    // ---- epilogue: optional bias, scatter C (16x16 f32 C/D layout) ----
    #pragma unroll
    for (int j = 0; j < 8; ++j) {
        int col  = j * 16 + lane15;
        float bv = bias ? bias[col] : 0.f;
        #pragma unroll
        for (int r = 0; r < 8; ++r) {
            int row = rowBase + wave * 16 + r + laneHi * 8;
            if (row < N) C[(size_t)row * DIMD + col] = acc[j][r] + bv;
        }
    }
}

// ---------------------------------------------------------------------------
// asrc[n] = h[n]·a_src ; adst[n] = h[n]·a_dst
// ---------------------------------------------------------------------------
__global__ void rowdot_kernel(const float* __restrict__ h,
                              const float* __restrict__ a_src,
                              const float* __restrict__ a_dst,
                              float* __restrict__ asrc, float* __restrict__ adst,
                              int N)
{
    int n = blockIdx.x * blockDim.x + threadIdx.x;
    if (n >= N) return;
    const float* hr = h + (size_t)n * DIMD;
    float s = 0.f, d = 0.f;
    #pragma unroll 8
    for (int i = 0; i < DIMD; ++i) {
        float v = hr[i];
        s += v * a_src[i];
        d += v * a_dst[i];
    }
    asrc[n] = s;
    adst[n] = d;
}

// ---------------------------------------------------------------------------
// init per-GAT scratch: emax keys to -inf, denom/agg to zero
// ---------------------------------------------------------------------------
__global__ void gat_init_kernel(unsigned* __restrict__ emax_key,
                                float* __restrict__ denom,
                                float* __restrict__ agg, int N)
{
    int i = blockIdx.x * blockDim.x + threadIdx.x;
    if (i < N) {
        emax_key[i] = KEY_NEG_INF;
        denom[i]    = 0.f;
    }
    if (i < N * DIMD) agg[i] = 0.f;
}

__device__ __forceinline__ void edge_endpoints(const int* __restrict__ ge,
                                               int E0, int e, int& s, int& d)
{
    if (e < E0) { s = ge[e]; d = ge[E0 + e]; }
    else        { s = d = e - E0; }          // appended self-loops
}

// pass 1: e = leaky_relu(asrc[s]+adst[d]); segment max via ordered-key atomic
__global__ void edge_pass1_kernel(const int* __restrict__ ge, int E0, int Etot,
                                  const float* __restrict__ asrc,
                                  const float* __restrict__ adst,
                                  float* __restrict__ eval,
                                  unsigned* __restrict__ emax_key)
{
    int e = blockIdx.x * blockDim.x + threadIdx.x;
    if (e >= Etot) return;
    int s, d; edge_endpoints(ge, E0, e, s, d);
    float v = asrc[s] + adst[d];
    v = (v > 0.f) ? v : 0.2f * v;            // NEG_SLOPE = 0.2
    eval[e] = v;
    atomicMax(&emax_key[d], f2key(v));
}

// pass 2: ex = exp(e - emax[d]); denom[d] += ex
__global__ void edge_pass2_kernel(const int* __restrict__ ge, int E0, int Etot,
                                  const unsigned* __restrict__ emax_key,
                                  float* __restrict__ eval,
                                  float* __restrict__ denom)
{
    int e = blockIdx.x * blockDim.x + threadIdx.x;
    if (e >= Etot) return;
    int s, d; edge_endpoints(ge, E0, e, s, d);
    float ex = __expf(eval[e] - key2f(emax_key[d]));
    eval[e] = ex;
    atomicAdd(&denom[d], ex);
}

// pass 3: agg[d] += (ex/denom[d]) * h[s]   (one wave per edge, float4/lane)
__global__ void edge_pass3_kernel(const int* __restrict__ ge, int E0, int Etot,
                                  const float* __restrict__ eval,
                                  const float* __restrict__ denom,
                                  const float* __restrict__ h,
                                  float* __restrict__ agg)
{
    int gid  = blockIdx.x * blockDim.x + threadIdx.x;
    int e    = gid >> 5;
    int lane = gid & 31;
    if (e >= Etot) return;
    int s, d; edge_endpoints(ge, E0, e, s, d);
    float coef = eval[e] / (denom[d] + 1e-16f);
    const float4 v = ((const float4*)(h + (size_t)s * DIMD))[lane];
    float* ad = agg + (size_t)d * DIMD + lane * 4;
    atomicAdd(ad + 0, coef * v.x);
    atomicAdd(ad + 1, coef * v.y);
    atomicAdd(ad + 2, coef * v.z);
    atomicAdd(ad + 3, coef * v.w);
}

// finalize: out = l2norm(agg + bias), one block (128 thr) per row
__global__ __launch_bounds__(128)
void gat_finalize_kernel(const float* __restrict__ agg,
                         const float* __restrict__ bias,
                         float* __restrict__ out, int N)
{
    __shared__ float red[DIMD];
    int n = blockIdx.x;
    int t = threadIdx.x;
    float v = agg[(size_t)n * DIMD + t] + bias[t];
    red[t] = v * v;
    __syncthreads();
    #pragma unroll
    for (int sft = 64; sft > 0; sft >>= 1) {
        if (t < sft) red[t] += red[t + sft];
        __syncthreads();
    }
    float nrm = fmaxf(sqrtf(red[0]), 1e-12f);
    out[(size_t)n * DIMD + t] = v / nrm;
}

// ---------------------------------------------------------------------------
// Host orchestration
// ---------------------------------------------------------------------------
extern "C" void kernel_launch(void* const* d_in, const int* in_sizes, int n_in,
                              void* d_out, int out_size, void* d_ws, size_t ws_size,
                              hipStream_t stream)
{
    // ---- inputs (setup_inputs order) ----
    const float* image_feat  = (const float*)d_in[1];
    const float* text_feat   = (const float*)d_in[2];
    const int*   ge          = (const int*)d_in[3];
    const float* entity_feat = (const float*)d_in[4];
    const float* W_img = (const float*)d_in[5];
    const float* b_img = (const float*)d_in[6];
    const float* W_txt = (const float*)d_in[7];
    const float* b_txt = (const float*)d_in[8];
    const float* W_fus = (const float*)d_in[9];
    const float* b_fus = (const float*)d_in[10];
    const float* Wg[3]   = { (const float*)d_in[11], (const float*)d_in[15], (const float*)d_in[19] };
    const float* aS[3]   = { (const float*)d_in[12], (const float*)d_in[16], (const float*)d_in[20] };
    const float* aD[3]   = { (const float*)d_in[13], (const float*)d_in[17], (const float*)d_in[21] };
    const float* bG[3]   = { (const float*)d_in[14], (const float*)d_in[18], (const float*)d_in[22] };

    const int N    = in_sizes[4] / DIMD;
    const int E0   = in_sizes[3] / 2;
    const int Etot = E0 + N;

    // ---- workspace carve-out (256B aligned) ----
    char*  base = (char*)d_ws;
    size_t off  = 0;
    auto carve = [&](size_t bytes) -> void* {
        off = (off + 255) & ~(size_t)255;
        void* p = base + off;
        off += bytes;
        return p;
    };
    float*    proj     = (float*)carve((size_t)N * DIMD * sizeof(float));
    float*    h        = (float*)carve((size_t)N * DIMD * sizeof(float));
    float*    agg      = (float*)carve((size_t)N * DIMD * sizeof(float));
    float*    asrc     = (float*)carve((size_t)N * sizeof(float));
    float*    adst     = (float*)carve((size_t)N * sizeof(float));
    float*    denom    = (float*)carve((size_t)N * sizeof(float));
    unsigned* emax_key = (unsigned*)carve((size_t)N * sizeof(unsigned));
    float*    eval     = (float*)carve((size_t)Etot * sizeof(float));
    (void)ws_size; (void)n_in; (void)out_size;

    const int gemmBlocks = (N + 127) / 128;

    auto run_gat = [&](const float* x, int mod, float* out) {
        // h = x @ Wg^T  (K = 128)
        gemm_bf16x3_kernel<<<gemmBlocks, 256, 0, stream>>>(
            x, DIMD, Wg[mod], DIMD, nullptr, h, N, DIMD, 0);
        rowdot_kernel<<<(N + 255) / 256, 256, 0, stream>>>(h, aS[mod], aD[mod], asrc, adst, N);
        gat_init_kernel<<<(N * DIMD + 255) / 256, 256, 0, stream>>>(emax_key, denom, agg, N);
        edge_pass1_kernel<<<(Etot + 255) / 256, 256, 0, stream>>>(ge, E0, Etot, asrc, adst, eval, emax_key);
        edge_pass2_kernel<<<(Etot + 255) / 256, 256, 0, stream>>>(ge, E0, Etot, emax_key, eval, denom);
        edge_pass3_kernel<<<(Etot * 32 + 255) / 256, 256, 0, stream>>>(ge, E0, Etot, eval, denom, h, agg);
        gat_finalize_kernel<<<N, 128, 0, stream>>>(agg, bG[mod], out, N);
    };

    float* out_e = (float*)d_out;
    float* out_i = out_e + (size_t)N * DIMD;
    float* out_t = out_i + (size_t)N * DIMD;
    float* out_m = out_t + (size_t)N * DIMD;

    // 1) entity modality
    run_gat(entity_feat, 0, out_e);

    // 2) image modality: proj = image_feat @ W_img^T + b_img  (K = 4096)
    gemm_bf16x3_kernel<<<gemmBlocks, 256, 0, stream>>>(
        image_feat, 4096, W_img, 4096, b_img, proj, N, 4096, 0);
    run_gat(proj, 1, out_i);

    // 3) text modality: proj = text_feat @ W_txt^T + b_txt  (K = 768)
    gemm_bf16x3_kernel<<<gemmBlocks, 256, 0, stream>>>(
        text_feat, 768, W_txt, 768, b_txt, proj, N, 768, 0);
    run_gat(proj, 2, out_t);

    // 4) fusion: mm = [e|i|t] @ W_fus^T + b_fus, as 3 accumulating K=128 passes
    gemm_bf16x3_kernel<<<gemmBlocks, 256, 0, stream>>>(
        out_e, DIMD, W_fus + 0,   3 * DIMD, nullptr, out_m, N, DIMD, 0);
    gemm_bf16x3_kernel<<<gemmBlocks, 256, 0, stream>>>(
        out_i, DIMD, W_fus + 128, 3 * DIMD, nullptr, out_m, N, DIMD, 1);
    gemm_bf16x3_kernel<<<gemmBlocks, 256, 0, stream>>>(
        out_t, DIMD, W_fus + 256, 3 * DIMD, b_fus,   out_m, N, DIMD, 1);
}
